// HSTUBlock_47304769798730
// MI455X (gfx1250) — matrix operand order repaired and checked by reading, not compiled, not run on hardware
//
#include <hip/hip_runtime.h>
#include <hip/hip_bf16.h>
#include <stdint.h>

typedef __attribute__((ext_vector_type(16))) __bf16 v16bf;
typedef __attribute__((ext_vector_type(8)))  __bf16 v8bf;
typedef __attribute__((ext_vector_type(8)))  float  v8f;

#define D_MODEL  1024
#define NUM_HEADS 16
#define HEAD_DIM  64
#define SEQ      2048
#define BATCH    2
#define MTOT     (BATCH*SEQ)   /* 4096 rows (tokens) */
#define N1       (4*D_MODEL)   /* 4096 proj cols: [u|v|q|k] */

__device__ __forceinline__ unsigned short f2bf(float f) {
  unsigned int u = __float_as_uint(f);
  u += 0x7FFFu + ((u >> 16) & 1u);       // round-to-nearest-even
  return (unsigned short)(u >> 16);
}
__device__ __forceinline__ float bf2f(unsigned short h) {
  return __uint_as_float(((unsigned int)h) << 16);
}
__device__ __forceinline__ float silu_f(float x) {
  return x / (1.0f + __expf(-x));
}
__device__ __forceinline__ v16bf cat8(v8bf lo, v8bf hi) {
  return __builtin_shufflevector(lo, hi, 0,1,2,3,4,5,6,7,8,9,10,11,12,13,14,15);
}
#define WMMA_BF16(A,B,C) \
  __builtin_amdgcn_wmma_f32_16x16x32_bf16(false,(A),false,(B),(short)0,(C),false,false)

// ---------------- conversion / transpose helpers ----------------

__global__ void k_cvt(const float* __restrict__ src, unsigned short* __restrict__ dst, int n) {
  int i = blockIdx.x * blockDim.x + threadIdx.x;
  if (i < n) dst[i] = f2bf(src[i]);
}

// dst[n*K + k] = src[k*N + n]   (row-major [K,N] fp32 -> N-major [N,K] bf16)
__global__ void k_cvt_t(const float* __restrict__ src, unsigned short* __restrict__ dst,
                        int Ncols, int Krows) {
  int i = blockIdx.x * blockDim.x + threadIdx.x;
  if (i < Ncols * Krows) {
    int n = i / Krows, k = i - n * Krows;
    dst[i] = f2bf(src[(size_t)k * Ncols + n]);
  }
}

// vt[(b*H+h)*64 + d][s] = proj[b*S+s][1024 + h*64 + d]   (bf16 copy)
__global__ void k_vt(const unsigned short* __restrict__ proj, unsigned short* __restrict__ vt) {
  int i = blockIdx.x * blockDim.x + threadIdx.x;  // 2^22 total
  if (i < BATCH * NUM_HEADS * HEAD_DIM * SEQ) {
    int s = i & (SEQ - 1);
    int d = (i >> 11) & (HEAD_DIM - 1);
    int h = (i >> 17) & (NUM_HEADS - 1);
    int b = i >> 21;
    vt[i] = proj[(size_t)(b * SEQ + s) * N1 + D_MODEL + h * HEAD_DIM + d];
  }
}

// ---------------- GEMM1: proj = silu(xb @ W1 + b1), bf16 out ----------------
// wave tile: 16(M) x 64(N); block = 4 waves -> 16 x 256

__global__ __launch_bounds__(128)
void k_gemm1(const unsigned short* __restrict__ xb,
             const unsigned short* __restrict__ w1t,   // [N1][D_MODEL]
             const float* __restrict__ b1,
             unsigned short* __restrict__ proj) {
  const int lane = threadIdx.x & 31;
  const int wave = threadIdx.x >> 5;
  const int l16 = lane & 15, half = lane >> 4;
  const int m0 = blockIdx.x * 16;
  const int n0 = blockIdx.y * 256 + wave * 64;

  const unsigned short* arow = xb + (size_t)(m0 + l16) * D_MODEL + half * 16;
  const unsigned short* br[4];
#pragma unroll
  for (int nt = 0; nt < 4; ++nt)
    br[nt] = w1t + (size_t)(n0 + nt * 16 + l16) * D_MODEL + half * 16;

  v8f acc[4] = {};
  for (int kb = 0; kb < D_MODEL; kb += 32) {
    __builtin_prefetch(arow + kb + 128, 0, 0);
    v16bf a = cat8(*(const v8bf*)(arow + kb), *(const v8bf*)(arow + kb + 8));
#pragma unroll
    for (int nt = 0; nt < 4; ++nt)
      acc[nt] = WMMA_BF16(a, *(const v16bf*)(br[nt] + kb), acc[nt]);
  }
#pragma unroll
  for (int nt = 0; nt < 4; ++nt) {
    int n = n0 + nt * 16 + l16;
    float bias = b1[n];
#pragma unroll
    for (int r = 0; r < 8; ++r) {
      int m = m0 + half * 8 + r;
      proj[(size_t)m * N1 + n] = f2bf(silu_f(acc[nt][r] + bias));
    }
  }
}

// ---------------- fused attention: av = (silu(QK^T)/S) @ V, gated by u ------
// one wave = one 16-row q tile of one (b,h); streams k in chunks of 32

__global__ __launch_bounds__(128)
void k_attn(const unsigned short* __restrict__ proj,
            const unsigned short* __restrict__ vt,     // [(b,h)*64 + d][s]
            float* __restrict__ g) {
  __shared__ __align__(32) unsigned short sm[4][16 * 32];
  const int lane = threadIdx.x & 31;
  const int wave = threadIdx.x >> 5;
  const int l16 = lane & 15, half = lane >> 4;
  const int tile = blockIdx.x * 4 + wave;
  const int q0 = (tile & 127) * 16;
  const int h = (tile >> 7) & (NUM_HEADS - 1);
  const int b = tile >> 11;

  // Q A-fragments (16x64 -> two 16x32 frags)
  const unsigned short* qrow =
      proj + (size_t)(b * SEQ + q0 + l16) * N1 + (2 * D_MODEL + h * HEAD_DIM);
  v16bf aq0 = cat8(*(const v8bf*)(qrow + half * 16),      *(const v8bf*)(qrow + half * 16 + 8));
  v16bf aq1 = cat8(*(const v8bf*)(qrow + 32 + half * 16), *(const v8bf*)(qrow + 40 + half * 16));

  const unsigned short* kbasep =
      proj + (size_t)(b * SEQ + l16) * N1 + (3 * D_MODEL + h * HEAD_DIM) + half * 16;
  const unsigned short* vbase =
      vt + (size_t)((b * NUM_HEADS + h) * HEAD_DIM + l16) * SEQ + half * 16;
  unsigned short* smw = sm[wave];

  v8f acc[4] = {};
  const float inv_s = 1.0f / (float)SEQ;

  for (int kb = 0; kb < SEQ; kb += 32) {
    // 16x32 score tile = two 16x16 WMMA C tiles (contract over d = 64)
#pragma unroll
    for (int jt = 0; jt < 2; ++jt) {
      const unsigned short* krow = kbasep + (size_t)(kb + jt * 16) * N1;
      v8f s = {};
      s = WMMA_BF16(aq0, *(const v16bf*)(krow), s);
      s = WMMA_BF16(aq1, *(const v16bf*)(krow + 32), s);
#pragma unroll
      for (int r = 0; r < 8; ++r) {
        int row = half * 8 + r;                        // C layout: M = r + 8*half
        smw[row * 32 + jt * 16 + l16] = f2bf(silu_f(s[r]) * inv_s);
      }
    }
    // C-layout -> A-layout transpose through LDS (same-wave, DS in-order)
    asm volatile("s_wait_dscnt 0" ::: "memory");
    v16bf aat = cat8(*(const v8bf*)(smw + l16 * 32 + half * 16),
                     *(const v8bf*)(smw + l16 * 32 + half * 16 + 8));
    // av += attn(16x32) @ V(32x64): 4 N-subtiles of 16
#pragma unroll
    for (int ds = 0; ds < 4; ++ds) {
      const unsigned short* vrow = vbase + (size_t)(ds * 16) * SEQ + kb;
      acc[ds] = WMMA_BF16(aat, *(const v16bf*)(vrow), acc[ds]);
    }
  }

  // gate with u (proj cols [0,1024)) and write fp32 g
#pragma unroll
  for (int ds = 0; ds < 4; ++ds) {
    int dcol = h * HEAD_DIM + ds * 16 + l16;
#pragma unroll
    for (int r = 0; r < 8; ++r) {
      int m = b * SEQ + q0 + half * 8 + r;
      float uval = bf2f(proj[(size_t)m * N1 + dcol]);
      g[(size_t)m * D_MODEL + dcol] = acc[ds][r] * uval;
    }
  }
}

// ---------------- LayerNorm: ln = LN(g)*gamma+beta, bf16 out ----------------

__global__ __launch_bounds__(256)
void k_ln(const float* __restrict__ g, const float* __restrict__ gamma,
          const float* __restrict__ beta, unsigned short* __restrict__ lnb) {
  __shared__ float s1[256], s2[256];
  const int row = blockIdx.x, t = threadIdx.x;
  const float* gr = g + (size_t)row * D_MODEL;
  float vals[4], s = 0.f, ss = 0.f;
#pragma unroll
  for (int i = 0; i < 4; ++i) {
    float v = gr[t + i * 256];
    vals[i] = v; s += v; ss += v * v;
  }
  s1[t] = s; s2[t] = ss;
  __syncthreads();
  for (int off = 128; off > 0; off >>= 1) {
    if (t < off) { s1[t] += s1[t + off]; s2[t] += s2[t + off]; }
    __syncthreads();
  }
  float mu = s1[0] * (1.0f / D_MODEL);
  float var = s2[0] * (1.0f / D_MODEL) - mu * mu;
  float rstd = rsqrtf(var + 1e-5f);
#pragma unroll
  for (int i = 0; i < 4; ++i) {
    int c = t + i * 256;
    lnb[(size_t)row * D_MODEL + c] = f2bf((vals[i] - mu) * rstd * gamma[c] + beta[c]);
  }
}

// ---------------- GEMM2: out = ln @ W2 + b2, fp32 out ----------------------

__global__ __launch_bounds__(128)
void k_gemm2(const unsigned short* __restrict__ lnb,
             const unsigned short* __restrict__ w2t,   // [D_MODEL][D_MODEL]
             const float* __restrict__ b2,
             float* __restrict__ out) {
  const int lane = threadIdx.x & 31;
  const int wave = threadIdx.x >> 5;
  const int l16 = lane & 15, half = lane >> 4;
  const int m0 = blockIdx.x * 16;
  const int n0 = blockIdx.y * 256 + wave * 64;

  const unsigned short* arow = lnb + (size_t)(m0 + l16) * D_MODEL + half * 16;
  const unsigned short* br[4];
#pragma unroll
  for (int nt = 0; nt < 4; ++nt)
    br[nt] = w2t + (size_t)(n0 + nt * 16 + l16) * D_MODEL + half * 16;

  v8f acc[4] = {};
  for (int kb = 0; kb < D_MODEL; kb += 32) {
    __builtin_prefetch(arow + kb + 128, 0, 0);
    v16bf a = cat8(*(const v8bf*)(arow + kb), *(const v8bf*)(arow + kb + 8));
#pragma unroll
    for (int nt = 0; nt < 4; ++nt)
      acc[nt] = WMMA_BF16(a, *(const v16bf*)(br[nt] + kb), acc[nt]);
  }
#pragma unroll
  for (int nt = 0; nt < 4; ++nt) {
    int n = n0 + nt * 16 + l16;
    float bias = b2[n];
#pragma unroll
    for (int r = 0; r < 8; ++r) {
      int m = m0 + half * 8 + r;
      out[(size_t)m * D_MODEL + n] = acc[nt][r] + bias;
    }
  }
}

// ---------------- launcher ----------------

extern "C" void kernel_launch(void* const* d_in, const int* in_sizes, int n_in,
                              void* d_out, int out_size, void* d_ws, size_t ws_size,
                              hipStream_t stream) {
  (void)in_sizes; (void)n_in; (void)out_size; (void)ws_size;
  const float* x     = (const float*)d_in[0];
  const float* W1    = (const float*)d_in[1];
  const float* b1    = (const float*)d_in[2];
  const float* W2    = (const float*)d_in[3];
  const float* b2    = (const float*)d_in[4];
  const float* gamma = (const float*)d_in[5];
  const float* beta  = (const float*)d_in[6];
  float* out = (float*)d_out;

  char* ws = (char*)d_ws;
  size_t o = 0;
  unsigned short* xb   = (unsigned short*)(ws + o); o += (size_t)MTOT * D_MODEL * 2;
  unsigned short* w1t  = (unsigned short*)(ws + o); o += (size_t)N1 * D_MODEL * 2;
  unsigned short* w2t  = (unsigned short*)(ws + o); o += (size_t)D_MODEL * D_MODEL * 2;
  unsigned short* proj = (unsigned short*)(ws + o); o += (size_t)MTOT * N1 * 2;
  unsigned short* vt   = (unsigned short*)(ws + o); o += (size_t)BATCH * NUM_HEADS * HEAD_DIM * SEQ * 2;
  float*          g    = (float*)(ws + o);          o += (size_t)MTOT * D_MODEL * 4;
  unsigned short* lnb  = (unsigned short*)(ws + o); o += (size_t)MTOT * D_MODEL * 2;

  const int nx = MTOT * D_MODEL;
  k_cvt   <<<(nx + 255) / 256, 256, 0, stream>>>(x, xb, nx);
  k_cvt_t <<<(N1 * D_MODEL + 255) / 256, 256, 0, stream>>>(W1, w1t, N1, D_MODEL);
  k_cvt_t <<<(D_MODEL * D_MODEL + 255) / 256, 256, 0, stream>>>(W2, w2t, D_MODEL, D_MODEL);

  k_gemm1 <<<dim3(MTOT / 16, N1 / 256), 128, 0, stream>>>(xb, w1t, b1, proj);
  k_vt    <<<(BATCH * NUM_HEADS * HEAD_DIM * SEQ + 255) / 256, 256, 0, stream>>>(proj, vt);
  k_attn  <<<(BATCH * NUM_HEADS * (SEQ / 16)) / 4, 128, 0, stream>>>(proj, vt, g);
  k_ln    <<<MTOT, 256, 0, stream>>>(g, gamma, beta, lnb);
  k_gemm2 <<<dim3(MTOT / 16, D_MODEL / 256), 128, 0, stream>>>(lnb, w2t, b2, out);
}
